// SingleChannelInterpolation_40707700031511
// MI455X (gfx1250) — matrix operand order repaired and checked by reading, not compiled, not run on hardware
//
#include <hip/hip_runtime.h>
#include <hip/hip_fp16.h>

typedef __attribute__((ext_vector_type(16))) _Float16 v16h;
typedef __attribute__((ext_vector_type(8)))  float    v8f;

#define B_ 32
#define C_ 128
#define W_ 128
#define H_ 128
#define KAPPA_ 10.0f
#define WAVES_PER_BLOCK 4
#define LOG2E_ 1.44269504088896340736f
#define LN2_   0.69314718055994530942f

// Per-wave private LDS region (waves are independent: one (b,c) pair each).
// All values in coef are pre-scaled by log2(e) so the inner loop is pure
// fma + v_exp_f32 (exp2), no per-entry ln<->log2 conversions.
struct WaveLds {
    float4    coef[W_];       // (a0, a1, a2, (1-kappa)*log2(m)) per w
    _Float16  xh[W_];         // x_t in f16 for the WMMA B fragment (col 0)
    _Float16  e[2][16][W_];   // exp tiles (both passes): 16 h-rows x 128 w, f16
    float     outbuf[2][16];  // [0]=numerator, [1]=denominator per local h
};

__launch_bounds__(WAVES_PER_BLOCK * 32)
__global__ void interp_wmma_kernel(const float* __restrict__ x_t,
                                   const float* __restrict__ d,
                                   const float* __restrict__ m,
                                   const float* __restrict__ kern,
                                   float* __restrict__ out)
{
    __shared__ WaveLds lds[WAVES_PER_BLOCK];

    const int lane = threadIdx.x & 31;
    const int wave = threadIdx.x >> 5;
    const int bc   = blockIdx.x * WAVES_PER_BLOCK + wave;   // 0 .. B*C-1
    WaveLds& L = lds[wave];

    const float* xb = x_t + bc * W_;
    const float* db = d   + bc * W_;
    const float* mb = m   + bc * W_;

    // ---- Stage 1: per-w coefficients (quadratic-in-t, log2 domain) --------
    // log2e*s(w,h) = (lm2 - alpha2*d^2) + (2*alpha2*d)*t + (-alpha2)*t^2
    //   alpha2 = log2e * softplus(kernel) = log2(1 + exp2(kernel*log2e))
    //   lm2    = log2(m)
    {
        const int w0 = lane * 4;                      // 4 w per lane, B128 loads
        float4 dv = *(const float4*)(db   + w0);
        float4 mv = *(const float4*)(mb   + w0);
        float4 xv = *(const float4*)(xb   + w0);
        float4 kv = *(const float4*)(kern + w0);
        float dd[4] = {dv.x, dv.y, dv.z, dv.w};
        float mm[4] = {mv.x, mv.y, mv.z, mv.w};
        float xx[4] = {xv.x, xv.y, xv.z, xv.w};
        float kk[4] = {kv.x, kv.y, kv.z, kv.w};
#pragma unroll
        for (int i = 0; i < 4; ++i) {
            float alpha2 = __builtin_amdgcn_logf(
                               1.0f + __builtin_amdgcn_exp2f(kk[i] * LOG2E_));
            float lm2 = __builtin_amdgcn_logf(mm[i]);   // m in [1e-3,1]: normal
            float a2 = -alpha2;
            float a1 = 2.0f * alpha2 * dd[i];
            float a0 = fmaf(-alpha2, dd[i] * dd[i], lm2);
            L.coef[w0 + i] = make_float4(a0, a1, a2, (1.0f - KAPPA_) * lm2);
            L.xh[w0 + i]   = (_Float16)xx[i];
        }
    }

    // ---- Hoisted B fragments: col 0 = x (numerator), col 1 = 1 (denom) ----
    // B col N = lane%16, element s holds K in table order; invariant over h-tiles/passes.
    v16h Bfrag[4];
#pragma unroll
    for (int kc = 0; kc < 4; ++kc) {
        union { v16h v; _Float16 h[16]; } Bf;
#pragma unroll
        for (int r = 0; r < 16; ++r) {
            const int kb = (r < 8) ? (8 * (lane >> 4) + r) : (16 + 8 * (lane >> 4) + (r - 8));
            Bf.h[r] = ((lane & 15) == 0) ? L.xh[kc * 32 + kb]   // same-wave LDS RAW: in-order DS pipe
                    : (((lane & 15) == 1) ? (_Float16)1.0f : (_Float16)0.0f);
        }
        Bfrag[kc] = Bf.v;
    }

    // lane -> (local h row, w-half) assignment for the score/exp stages
    const int   hl    = lane & 15;     // local h row 0..15
    const int   whalf = lane >> 4;     // 0: w in [0,64), 1: w in [64,128)
    const float step  = (float)H_ / (float)(H_ - 1);   // linspace(0,H,H)

    for (int ht = 0; ht < H_ / 16; ++ht) {
        const float t = (float)(ht * 16 + hl) * step;

        // ---- Per-h running max for both kernels (logsumexp, log2 domain) --
        float mx0 = -3.0e38f, mx1 = -3.0e38f;
#pragma unroll 8
        for (int j = 0; j < 64; ++j) {
            float4 c = L.coef[whalf * 64 + j];
            float s  = fmaf(fmaf(c.z, t, c.y), t, c.x);
            float st = fmaf(KAPPA_, s, c.w);          // c.w = (1-k)*log2(m)
            mx0 = fmaxf(mx0, s);
            mx1 = fmaxf(mx1, st);
        }
        // combine the two w-halves (lane L <-> lane L^16)
        mx0 = fmaxf(mx0, __shfl_xor(mx0, 16, 32));
        mx1 = fmaxf(mx1, __shfl_xor(mx1, 16, 32));

        // ---- Fused exp-tile build for BOTH kernels: coef loaded once, -----
        // s computed once, s_t = fma(kappa, s, c.w). Packed 2-wide f16 stores.
#pragma unroll 4
        for (int j = 0; j < 64; j += 2) {
            const int w = whalf * 64 + j;
            float4 c0 = L.coef[w];
            float4 c1 = L.coef[w + 1];
            float s0  = fmaf(fmaf(c0.z, t, c0.y), t, c0.x);
            float s1  = fmaf(fmaf(c1.z, t, c1.y), t, c1.x);
            float st0 = fmaf(KAPPA_, s0, c0.w);
            float st1 = fmaf(KAPPA_, s1, c1.w);
            union { _Float16 h[2]; unsigned u; } p0, p1;
            p0.h[0] = (_Float16)__builtin_amdgcn_exp2f(s0  - mx0);
            p0.h[1] = (_Float16)__builtin_amdgcn_exp2f(s1  - mx0);
            p1.h[0] = (_Float16)__builtin_amdgcn_exp2f(st0 - mx1);
            p1.h[1] = (_Float16)__builtin_amdgcn_exp2f(st1 - mx1);
            *(unsigned*)&L.e[0][hl][w] = p0.u;
            *(unsigned*)&L.e[1][hl][w] = p1.u;
        }

        // ---- Two WMMA chains: D(16h x 16n) = sum_K A(16h x 32w)*B(32w x 16n)
        for (int p = 0; p < 2; ++p) {
            const int M  = lane & 15;        // A row (local h)
            const int kh = lane >> 4;        // K-half select per 16-bit A layout

            // Preload ALL A fragments so the 4 WMMAs issue back-to-back
            // (no per-WMMA s_wait_dscnt 0 drain).
            union { v16h v; unsigned u[8]; } A[4];
#pragma unroll
            for (int kc = 0; kc < 4; ++kc) {
#pragma unroll
                for (int r = 0; r < 8; ++r) {
                    // 16-bit A layout: VGPR r holds K = (r/4)*16 + kh*8 + (r%4)*2 {+0,+1}
                    int kp = ((r >> 2) << 4) + kh * 8 + ((r & 3) << 1);
                    A[kc].u[r] = *(const unsigned*)&L.e[p][M][kc * 32 + kp];
                }
            }
            v8f acc = {};
#pragma unroll
            for (int kc = 0; kc < 4; ++kc)   // K = 128 in 4 chunks of 32
                acc = __builtin_amdgcn_wmma_f32_16x16x32_f16(
                    false, A[kc].v, false, Bfrag[kc], (short)0, acc, false, false);

            // ---- Extract N=0 (numerator) and N=1 (denominator) columns -----
            {
                const int N     = lane & 15;
                const int mbase = (lane >> 4) * 8;   // C layout: lanes 16-31 hold M+8
                if (N < 2) {
#pragma unroll
                    for (int r = 0; r < 8; ++r)
                        L.outbuf[N][mbase + r] = acc[r];
                }
            }

            if (lane < 16) {
                const int   hg  = ht * 16 + lane;
                const float num = L.outbuf[0][lane];
                const float den = L.outbuf[1][lane];
                const float y   = num * __builtin_amdgcn_rcpf(den);
                if (p == 0) {
                    out[bc * H_ + hg] = y;                               // y
                    // w (natural-log lse) = ln2 * (mx2 + log2(den))
                    out[B_ * C_ * H_ + bc * H_ + hg] =
                        fmaf(LN2_, __builtin_amdgcn_logf(den), LN2_ * mx0);
                } else {
                    out[2 * B_ * C_ * H_ + bc * H_ + hg] = y;            // y_trans
                }
            }
        }
    }
}

extern "C" void kernel_launch(void* const* d_in, const int* in_sizes, int n_in,
                              void* d_out, int out_size, void* d_ws, size_t ws_size,
                              hipStream_t stream) {
    (void)in_sizes; (void)n_in; (void)out_size; (void)d_ws; (void)ws_size;
    const float* x_t  = (const float*)d_in[0];
    const float* d    = (const float*)d_in[1];
    const float* m    = (const float*)d_in[2];
    const float* kern = (const float*)d_in[3];
    float* out = (float*)d_out;

    const int blocks = (B_ * C_) / WAVES_PER_BLOCK;   // 4096 (b,c) pairs, 4/block
    interp_wmma_kernel<<<blocks, WAVES_PER_BLOCK * 32, 0, stream>>>(
        x_t, d, m, kern, out);
}
